// Attention_17944373362766
// MI455X (gfx1250) — compile-verified, matrix operand order checked by reading, not run on hardware
//
#include <hip/hip_runtime.h>
#include <hip/hip_bf16.h>

// ---------------------------------------------------------------------------
// Fused attention for B=2, S=2048, D=1024, H=16, HD=64 on gfx1250 (MI455X).
// bf16 WMMA (V_WMMA_F32_16X16X32_BF16) everywhere, f32 accumulation,
// f32 online softmax.  Tile feeds use GLOBAL_LOAD_ASYNC_TO_LDS_B128
// (ASYNCcnt-tracked LDS DMA) with double buffering in the GEMM.
// ---------------------------------------------------------------------------

typedef __attribute__((ext_vector_type(16))) __bf16 v16bf;
typedef __attribute__((ext_vector_type(8)))  float  v8f;

#define S_LEN 2048
#define DIM   1024
#define NH    16
#define HD    64

__device__ __forceinline__ v8f wmma_bf16(v16bf a, v16bf b, v8f c) {
    // D = A(16x32 bf16) * B(32x16 bf16) + C(16x16 f32)
    return __builtin_amdgcn_wmma_f32_16x16x32_bf16(
        /*neg_a=*/false, a, /*neg_b=*/false, b,
        /*c_mod=*/(short)0, c, /*reuse_a=*/false, /*reuse_b=*/false);
}

// Async DMA: 16 bytes global -> LDS, tracked by ASYNCcnt (no VGPR data path).
__device__ __forceinline__ void async_copy_b128(void* lds_ptr, const void* gptr) {
    uint32_t lds_off = (uint32_t)(uintptr_t)lds_ptr;   // LDS aperture: low 32 bits = LDS byte offset
    uint64_t gaddr   = (uint64_t)(uintptr_t)gptr;
    asm volatile("global_load_async_to_lds_b128 %0, %1, off"
                 :: "v"(lds_off), "v"(gaddr)
                 : "memory");
}

__device__ __forceinline__ void wait_asynccnt0() {
    asm volatile("s_wait_asynccnt 0x0" ::: "memory");
}

// ---------------------------------------------------------------------------
// Kernel 0: f32 -> bf16 conversion (x and the three weight matrices).
// ---------------------------------------------------------------------------
__global__ __launch_bounds__(256) void f32_to_bf16_kernel(
    const float* __restrict__ src, __bf16* __restrict__ dst, int n)
{
    int i = blockIdx.x * blockDim.x + threadIdx.x;
    if (i < n) dst[i] = (__bf16)src[i];
}

// ---------------------------------------------------------------------------
// Kernel 1: QKV projection (x @ W^T + bias) with fused RoPE on Q,K.
// Tile: 128(M) x 128(N), K-step 32.  Block = 256 threads = 8 waves,
// waves arranged 2(M) x 4(N); each wave computes 64x32 via 4x2 WMMA accs.
// Tiles stream in through async LDS DMA, double buffered so the DMA of
// tile k+1 overlaps the 8 WMMAs of tile k.
// Output layout: [B, H, S, HD] bf16.  blockIdx.z selects Q/K/V.
// ---------------------------------------------------------------------------
__global__ __launch_bounds__(256) void qkv_rope_gemm(
    const __bf16* __restrict__ xb,
    const __bf16* __restrict__ wqb, const __bf16* __restrict__ wkb,
    const __bf16* __restrict__ wvb,
    const float*  __restrict__ bq,  const float*  __restrict__ bk,
    const float*  __restrict__ bv,
    __bf16* __restrict__ Qt, __bf16* __restrict__ Kt, __bf16* __restrict__ Vt)
{
    const int z = blockIdx.z;
    const __bf16* W    = (z == 0) ? wqb : (z == 1) ? wkb : wvb;
    const float*  bias = (z == 0) ? bq  : (z == 1) ? bk  : bv;
    __bf16*       dst  = (z == 0) ? Qt  : (z == 1) ? Kt  : Vt;

    const int tileM = blockIdx.x * 128;
    const int tileN = blockIdx.y * 128;
    const int tid   = threadIdx.x;
    const int lane  = tid & 31;
    const int wid   = tid >> 5;
    const int wm    = (wid & 1) * 64;   // wave M offset inside tile
    const int wn    = (wid >> 1) * 32;  // wave N offset inside tile
    const int lm    = lane & 15;        // N / M stripe index
    const int lh    = lane >> 4;        // K-half select

    __shared__ __align__(32) __bf16 As[2][128 * 32];
    __shared__ __align__(32) __bf16 Bs[2][128 * 32];

    // This thread's 32B slice of each tile (two b128 DMAs per matrix).
    const int r  = tid >> 1;
    const int hf = (tid & 1) * 16;
    const __bf16* gA = &xb[(size_t)(tileM + r) * DIM + hf];
    const __bf16* gB = &W [(size_t)(tileN + r) * DIM + hf];

    v8f acc[4][2];
#pragma unroll
    for (int mi = 0; mi < 4; ++mi)
#pragma unroll
        for (int ni = 0; ni < 2; ++ni) acc[mi][ni] = (v8f){};

    // Prologue: DMA tile k0=0 into buffer 0.
    async_copy_b128(&As[0][r * 32 + hf],     gA);
    async_copy_b128(&As[0][r * 32 + hf + 8], gA + 8);
    async_copy_b128(&Bs[0][r * 32 + hf],     gB);
    async_copy_b128(&Bs[0][r * 32 + hf + 8], gB + 8);

    int cur = 0;
    for (int k0 = 0; k0 < DIM; k0 += 32) {
        wait_asynccnt0();        // my DMA slices for buffer `cur` are in LDS
        __syncthreads();         // everyone's slices are in LDS

        if (k0 + 32 < DIM) {     // kick off DMA for the next tile
            const __bf16* nA = gA + (k0 + 32);
            const __bf16* nB = gB + (k0 + 32);
            int nb = cur ^ 1;
            async_copy_b128(&As[nb][r * 32 + hf],     nA);
            async_copy_b128(&As[nb][r * 32 + hf + 8], nA + 8);
            async_copy_b128(&Bs[nb][r * 32 + hf],     nB);
            async_copy_b128(&Bs[nb][r * 32 + hf + 8], nB + 8);
        }

        v16bf bfrag[2];
#pragma unroll
        for (int ni = 0; ni < 2; ++ni)
            bfrag[ni] = *(v16bf*)&Bs[cur][(wn + ni * 16 + lm) * 32 + lh * 16];
#pragma unroll
        for (int mi = 0; mi < 4; ++mi) {
            v16bf afrag = *(v16bf*)&As[cur][(wm + mi * 16 + lm) * 32 + lh * 16];
#pragma unroll
            for (int ni = 0; ni < 2; ++ni)
                acc[mi][ni] = wmma_bf16(afrag, bfrag[ni], acc[mi][ni]);
        }
        cur ^= 1;
    }

    // Epilogue: bias, RoPE (Q,K only), scatter to [B,H,S,HD] bf16.
    const float log2_10k_over_32 = 0.41524101186092029f; // log2(10000)/32
#pragma unroll
    for (int mi = 0; mi < 4; ++mi) {
#pragma unroll
        for (int ni = 0; ni < 2; ++ni) {
#pragma unroll
            for (int v = 0; v < 8; ++v) {
                int m = tileM + wm + mi * 16 + v + ((lane >= 16) ? 8 : 0);
                int n = tileN + wn + ni * 16 + lm;
                float val = acc[mi][ni][v] + bias[n];
                if (z < 2) {
                    // RoPE: pair partner lives one lane over (n ^ 1).
                    float prt = __shfl_xor(val, 1, 32);
                    int   s   = m & (S_LEN - 1);
                    int   hd  = n & (HD - 1);
                    int   j   = hd >> 1;
                    float ang = (float)s * exp2f(-(float)j * log2_10k_over_32);
                    float c   = __cosf(ang);
                    float sn  = __sinf(ang);
                    val = (hd & 1) ? (prt * sn + val * c)
                                   : (val * c - prt * sn);
                }
                int b = m >> 11, s = m & (S_LEN - 1);
                int h = n >> 6,  hd = n & (HD - 1);
                dst[((((size_t)b * NH + h) * S_LEN) + s) * HD + hd] = (__bf16)val;
            }
        }
    }
}

// ---------------------------------------------------------------------------
// Kernel 2: flash attention.  Block = 128 threads = 4 waves; each wave owns a
// 16-row Q stripe (block covers 64 Q rows of one (b,h)).  KV streamed in
// 32-row tiles: K via async LDS DMA, V load+transpose through VGPRs so the
// P@V B-fragments are contiguous 32B LDS reads.
// ---------------------------------------------------------------------------
__global__ __launch_bounds__(128) void flash_attn(
    const __bf16* __restrict__ Qt, const __bf16* __restrict__ Kt,
    const __bf16* __restrict__ Vt, float* __restrict__ out)
{
    const int bh   = blockIdx.y;            // b*H + h
    const int b    = bh >> 4;
    const int h    = bh & 15;
    const int tid  = threadIdx.x;
    const int lane = tid & 31;
    const int wid  = tid >> 5;
    const int lm   = lane & 15;
    const int lh   = lane >> 4;

    const size_t head_off = ((size_t)bh) * S_LEN * HD;
    const __bf16* Qh = Qt + head_off;
    const __bf16* Kh = Kt + head_off;
    const __bf16* Vh = Vt + head_off;

    __shared__ __align__(32) __bf16 Ks[32 * 64];      // [kv][hd]
    __shared__ __align__(32) __bf16 Vs[64 * 32];      // [hd][kv] (transposed)
    __shared__ __align__(32) __bf16 Ps[4][16 * 32];   // per-wave P tile

    const int qrow0 = blockIdx.x * 64 + wid * 16;

    // Q A-fragments: 16x64, split into two 16x32 fragments.
    const int qs = qrow0 + lm;
    v16bf qa0 = *(const v16bf*)&Qh[(size_t)qs * HD + lh * 16];
    v16bf qa1 = *(const v16bf*)&Qh[(size_t)qs * HD + 32 + lh * 16];

    v8f o[4];
#pragma unroll
    for (int n = 0; n < 4; ++n) o[n] = (v8f){};
    float rmax[8], rsum[8];
#pragma unroll
    for (int v = 0; v < 8; ++v) { rmax[v] = -__builtin_inff(); rsum[v] = 0.f; }

    const int r  = tid >> 2;            // kv row 0..31
    const int c0 = (tid & 3) * 16;      // hd chunk

    for (int kv0 = 0; kv0 < S_LEN; kv0 += 32) {
        __syncthreads();                // previous tile fully consumed
        {   // K tile: async DMA straight into LDS
            const __bf16* gK = &Kh[(size_t)(kv0 + r) * HD + c0];
            async_copy_b128(&Ks[r * 64 + c0],     gK);
            async_copy_b128(&Ks[r * 64 + c0 + 8], gK + 8);
            // V tile: load + transpose through VGPRs
            v16bf vt = *(const v16bf*)&Vh[(size_t)(kv0 + r) * HD + c0];
#pragma unroll
            for (int j = 0; j < 16; ++j) Vs[(c0 + j) * 32 + r] = vt[j];
        }
        wait_asynccnt0();
        __syncthreads();

        // scores: S = Q(16x64) @ K_tile^T(64x32), two 16x16 chunks
        v8f s0 = (v8f){}, s1 = (v8f){};
#pragma unroll
        for (int kc = 0; kc < 2; ++kc) {
            v16bf qa = kc ? qa1 : qa0;
            v16bf kb0 = *(v16bf*)&Ks[lm * 64 + kc * 32 + lh * 16];
            v16bf kb1 = *(v16bf*)&Ks[(16 + lm) * 64 + kc * 32 + lh * 16];
            s0 = wmma_bf16(qa, kb0, s0);
            s1 = wmma_bf16(qa, kb1, s1);
        }

        // online softmax (per row, rows striped: M = v + 8*(lane>=16))
        float corr[8];
#pragma unroll
        for (int v = 0; v < 8; ++v) {
            float a  = s0[v] * 0.125f;          // 1/sqrt(64)
            float c2 = s1[v] * 0.125f;
            float mx = fmaxf(a, c2);
            mx = fmaxf(mx, __shfl_xor(mx, 1, 32));
            mx = fmaxf(mx, __shfl_xor(mx, 2, 32));
            mx = fmaxf(mx, __shfl_xor(mx, 4, 32));
            mx = fmaxf(mx, __shfl_xor(mx, 8, 32));
            float nm = fmaxf(rmax[v], mx);
            corr[v]  = __expf(rmax[v] - nm);
            rmax[v]  = nm;
            float p0 = __expf(a - nm);
            float p1 = __expf(c2 - nm);
            float rs = p0 + p1;
            rs += __shfl_xor(rs, 1, 32);
            rs += __shfl_xor(rs, 2, 32);
            rs += __shfl_xor(rs, 4, 32);
            rs += __shfl_xor(rs, 8, 32);
            rsum[v] = rsum[v] * corr[v] + rs;
            int ml = v + ((lane >= 16) ? 8 : 0);
            Ps[wid][ml * 32 + lm]      = (__bf16)p0;
            Ps[wid][ml * 32 + 16 + lm] = (__bf16)p1;
        }
#pragma unroll
        for (int n = 0; n < 4; ++n)
#pragma unroll
            for (int v = 0; v < 8; ++v) o[n][v] *= corr[v];

        // wave-private LDS: make P stores visible to the re-layout loads
        asm volatile("s_wait_dscnt 0x0" ::: "memory");

        // O += P(16x32) @ V_tile(32x64)
        v16bf pa = *(v16bf*)&Ps[wid][lm * 32 + lh * 16];
#pragma unroll
        for (int n = 0; n < 4; ++n) {
            v16bf vb = *(v16bf*)&Vs[(n * 16 + lm) * 32 + lh * 16];
            o[n] = wmma_bf16(pa, vb, o[n]);
        }
    }

    // epilogue: normalize and store f32 [B, S, D] (transposed back)
#pragma unroll
    for (int n = 0; n < 4; ++n) {
#pragma unroll
        for (int v = 0; v < 8; ++v) {
            int ml   = v + ((lane >= 16) ? 8 : 0);
            int srow = qrow0 + ml;
            int hd   = n * 16 + lm;
            out[((size_t)b * S_LEN + srow) * DIM + h * HD + hd] =
                o[n][v] / rsum[v];
        }
    }
}

// ---------------------------------------------------------------------------
// Launch
// ---------------------------------------------------------------------------
extern "C" void kernel_launch(void* const* d_in, const int* in_sizes, int n_in,
                              void* d_out, int out_size, void* d_ws, size_t ws_size,
                              hipStream_t stream) {
    const float* x    = (const float*)d_in[0];
    const float* wq_w = (const float*)d_in[1];
    const float* wq_b = (const float*)d_in[2];
    const float* wk_w = (const float*)d_in[3];
    const float* wk_b = (const float*)d_in[4];
    const float* wv_w = (const float*)d_in[5];
    const float* wv_b = (const float*)d_in[6];
    float* out = (float*)d_out;

    char* ws = (char*)d_ws;
    const size_t MB = (size_t)1 << 20;
    __bf16* xb  = (__bf16*)(ws);             //  8 MB  x bf16 [4096][1024]
    __bf16* wqb = (__bf16*)(ws +  8 * MB);   //  2 MB
    __bf16* wkb = (__bf16*)(ws + 10 * MB);   //  2 MB
    __bf16* wvb = (__bf16*)(ws + 12 * MB);   //  2 MB
    __bf16* Qt  = (__bf16*)(ws + 14 * MB);   //  8 MB  [B,H,S,HD]
    __bf16* Kt  = (__bf16*)(ws + 22 * MB);   //  8 MB
    __bf16* Vt  = (__bf16*)(ws + 30 * MB);   //  8 MB  (38 MB total)

    f32_to_bf16_kernel<<<16384, 256, 0, stream>>>(x, xb, 4096 * 1024);
    f32_to_bf16_kernel<<<4096, 256, 0, stream>>>(wq_w, wqb, 1024 * 1024);
    f32_to_bf16_kernel<<<4096, 256, 0, stream>>>(wk_w, wkb, 1024 * 1024);
    f32_to_bf16_kernel<<<4096, 256, 0, stream>>>(wv_w, wvb, 1024 * 1024);

    dim3 g1(4096 / 128, 1024 / 128, 3);   // (32, 8, 3)
    qkv_rope_gemm<<<g1, 256, 0, stream>>>(xb, wqb, wkb, wvb,
                                          wq_b, wk_b, wv_b, Qt, Kt, Vt);

    dim3 g2(2048 / 64, 2 * 16);           // (32, 32)
    flash_attn<<<g2, 128, 0, stream>>>(Qt, Kt, Vt, out);
}
